// GAT_18915035971953
// MI455X (gfx1250) — matrix operand order repaired and checked by reading, not compiled, not run on hardware
//
#include <hip/hip_runtime.h>
#include <hip/hip_bf16.h>
#include <math.h>

// ---------------------------------------------------------------------------
// 2-layer GAT (PyG GATConv, concat=False) for MI455X / gfx1250.
// Dense projections via v_wmma_f32_16x16x32_bf16 (one 16x16 C tile per wave32),
// with fully vectorized (b128) fragment loads: A row-major bf16, B pre-
// transposed to [N][K] bf16 so both operands are contiguous per lane.
// Edge softmax: 3-pass (atomic max -> exp+atomic sum -> weighted scatter-add),
// with the head-mean folded into the scatter so [E,H,F] never materializes.
// ---------------------------------------------------------------------------

typedef __attribute__((ext_vector_type(16))) __bf16          v16bf;
typedef __attribute__((ext_vector_type(8)))  float           v8f;
typedef __attribute__((ext_vector_type(8)))  unsigned short  us8;

__device__ __forceinline__ unsigned short f32_to_bf16_rne(float f) {
  unsigned u = __float_as_uint(f);
  unsigned r = u + 0x7FFFu + ((u >> 16) & 1u);   // round-to-nearest-even
  return (unsigned short)(r >> 16);
}
// Order-preserving float<->uint encoding for atomicMax over signed floats.
__device__ __forceinline__ unsigned fenc(float f) {
  unsigned u = __float_as_uint(f);
  return (u & 0x80000000u) ? ~u : (u | 0x80000000u);
}
__device__ __forceinline__ float fdec(unsigned u) {
  unsigned v = (u & 0x80000000u) ? (u & 0x7FFFFFFFu) : ~u;
  return __uint_as_float(v);
}
// Host-side constant: fenc(-inf) = ~bits(-inf) = ~0xFF800000 = 0x007FFFFF.
#define ENC_NEG_INF_U32 0x007FFFFFu

// ---------------- utility kernels ----------------

__global__ void k_fill_u32(unsigned* p, unsigned v, int n) {
  int i = blockIdx.x * blockDim.x + threadIdx.x;
  if (i < n) p[i] = v;
}

__global__ void k_f32_to_bf16(const float* in, unsigned short* out, int n) {
  int i = blockIdx.x * blockDim.x + threadIdx.x;
  if (i < n) out[i] = f32_to_bf16_rne(in[i]);
}

// Transpose + cast: in[K,N] f32 (row-major) -> out[N,K] bf16. K*N is tiny
// (<= 32768 elements), so a naive thread-per-element kernel is fine.
__global__ void k_transpose_f32_to_bf16(const float* __restrict__ in,
                                        unsigned short* __restrict__ out,
                                        int K, int N) {
  int i = blockIdx.x * blockDim.x + threadIdx.x;
  if (i >= K * N) return;
  const int k = i / N, n = i - k * N;
  out[(size_t)n * K + k] = f32_to_bf16_rne(in[i]);
}

// ---------------- WMMA GEMM: C[M,N] = A[M,K](bf16,rm) * Bt[N,K](bf16) --------
// block = (32,8): 8 waves; wave w computes tile (blockIdx.x, blockIdx.y*8+w).
// K is a compile-time constant (128 or 32) so the K-loop fully unrolls and
// all fragment loads are 16-byte global_load_b128.

template <int K>
__global__ void k_wmma_gemm_bf16(const unsigned short* __restrict__ A,
                                 const unsigned short* __restrict__ Bt,
                                 float* __restrict__ C, int M, int N) {
  const int lane  = threadIdx.x;              // 0..31
  const int wave  = threadIdx.y;              // 0..7
  const int tileM = blockIdx.x;
  const int tileN = blockIdx.y * 8 + wave;
  if (tileN * 16 >= N) return;                // wave-uniform; EXEC stays all-1s

  const int l15   = lane & 15;
  const int khalf = lane >> 4;                // 0 or 1
  const int row   = tileM * 16 + l15;         // A row held by this lane
  const int col   = tileN * 16 + l15;         // B/C column held by this lane

  const unsigned short* arow = A  + (size_t)row * K;   // 16B-aligned (K%32==0)
  const unsigned short* bcol = Bt + (size_t)col * K;

  v8f acc = {};
#pragma unroll
  for (int kb = 0; kb < K; kb += 32) {
    // A 16x32 bf16 layout (ISA 7.12.2): lane half selects interleaved K blocks
    // of 8; each block is contiguous in memory -> one b128 load per block.
    const us8 a0 = *(const us8*)(arow + kb + (khalf << 3));        // K=kb+kh*8+0..7
    const us8 a1 = *(const us8*)(arow + kb + 16 + (khalf << 3));   // +16
    // B 32x16 bf16: same interleaved K blocks of 8 as A, contiguous in Bt.
    const us8 b0 = *(const us8*)(bcol + kb + (khalf << 3));
    const us8 b1 = *(const us8*)(bcol + kb + 16 + (khalf << 3));
    v16bf a, b;
    __builtin_memcpy(&a, &a0, 16); __builtin_memcpy((char*)&a + 16, &a1, 16);
    __builtin_memcpy(&b, &b0, 16); __builtin_memcpy((char*)&b + 16, &b1, 16);
    acc = __builtin_amdgcn_wmma_f32_16x16x32_bf16(
        /*neg_a=*/false, a, /*neg_b=*/false, b,
        /*c_mod=*/(short)0, acc, /*reuse_a=*/false, /*reuse_b=*/false);
  }
#pragma unroll
  for (int j = 0; j < 8; ++j) {               // C: VGPR j -> M = 8*khalf + j
    const int r = tileM * 16 + (khalf << 3) + j;
    C[(size_t)r * N + col] = acc[j];
  }
}

// ---------------- attention coefficients ----------------
// a_s[n,h] = <h[n,h,:], att_src[h,:]> ; a_d likewise. One thread per (n,h).

__global__ void k_att_coef(const float* __restrict__ h,
                           const float* __restrict__ att_src,
                           const float* __restrict__ att_dst,
                           float* __restrict__ a_s, float* __restrict__ a_d,
                           int Nn, int H, int F) {
  int i = blockIdx.x * blockDim.x + threadIdx.x;
  if (i >= Nn * H) return;
  const int n = i / H, hh = i - n * H;
  const float* hv = h + (size_t)n * H * F + (size_t)hh * F;
  const float* ps = att_src + hh * F;
  const float* pd = att_dst + hh * F;
  float vs = 0.f, vd = 0.f;
  for (int f = 0; f < F; ++f) { const float x = hv[f]; vs += x * ps[f]; vd += x * pd[f]; }
  a_s[i] = vs; a_d[i] = vd;
}

// ---------------- edge softmax pass 1: segment max ----------------

__global__ void k_edge_max(const int* __restrict__ src, const int* __restrict__ dst,
                           const float* __restrict__ a_s, const float* __restrict__ a_d,
                           unsigned* __restrict__ menc, int E, int H) {
  int i = blockIdx.x * blockDim.x + threadIdx.x;
  if (i >= E * H) return;
  const int e = i / H, hh = i - e * H;
  const int s = src[e], d = dst[e];
  float v = a_s[s * H + hh] + a_d[d * H + hh];
  v = (v > 0.f) ? v : 0.2f * v;               // leaky_relu(0.2)
  atomicMax(&menc[d * H + hh], fenc(v));
}

// ---------------- edge softmax pass 2: exp + segment sum ----------------

__global__ void k_edge_expsum(const int* __restrict__ src, const int* __restrict__ dst,
                              const float* __restrict__ a_s, const float* __restrict__ a_d,
                              const unsigned* __restrict__ menc,
                              float* __restrict__ alpha_num, float* __restrict__ ssum,
                              int E, int H) {
  int i = blockIdx.x * blockDim.x + threadIdx.x;
  if (i >= E * H) return;
  const int e = i / H, hh = i - e * H;
  const int s = src[e], d = dst[e];
  float v = a_s[s * H + hh] + a_d[d * H + hh];
  v = (v > 0.f) ? v : 0.2f * v;
  const float m = fdec(menc[d * H + hh]);
  const float a = __expf(v - m);
  alpha_num[i] = a;
  atomicAdd(&ssum[d * H + hh], a);
}

// ---------------- edge softmax pass 3: weighted scatter-add ----------------
// Folds 1/H head-mean into the accumulation: acc[d,f] += alpha/H * h[s,hh,f].

__global__ void k_message(const int* __restrict__ src, const int* __restrict__ dst,
                          const float* __restrict__ h,
                          const float* __restrict__ alpha_num,
                          const float* __restrict__ ssum,
                          float* __restrict__ acc, int E, int H, int F, float invH) {
  int i = blockIdx.x * blockDim.x + threadIdx.x;
  if (i >= E * H) return;
  const int e = i / H, hh = i - e * H;
  const int s = src[e], d = dst[e];
  const float alpha = alpha_num[i] / (ssum[d * H + hh] + 1e-16f) * invH;
  const float* hv = h + (size_t)s * H * F + (size_t)hh * F;
  float* av = acc + (size_t)d * F;
  for (int f = 0; f < F; ++f) atomicAdd(&av[f], alpha * hv[f]);
}

// ---------------- layer-1 epilogue: +bias, relu, cast to bf16 ----------------

__global__ void k_finalize1(const float* __restrict__ acc, const float* __restrict__ b1,
                            unsigned short* __restrict__ hmid_b, int Nn, int F) {
  int i = blockIdx.x * blockDim.x + threadIdx.x;
  if (i >= Nn * F) return;
  float v = acc[i] + b1[i % F];
  v = (v > 0.f) ? v : 0.f;
  hmid_b[i] = f32_to_bf16_rne(v);
}

// ---------------- layer-2 epilogue: +bias (in place on d_out) ----------------

__global__ void k_finalize2(float* __restrict__ out, const float* __restrict__ b2,
                            int Nn, int F) {
  int i = blockIdx.x * blockDim.x + threadIdx.x;
  if (i >= Nn * F) return;
  out[i] += b2[i % F];
}

// ---------------------------------------------------------------------------

extern "C" void kernel_launch(void* const* d_in, const int* in_sizes, int n_in,
                              void* d_out, int out_size, void* d_ws, size_t ws_size,
                              hipStream_t stream) {
  (void)n_in; (void)ws_size; (void)out_size;
  const int IN_F = 128, H = 8, HF = 32, OUT_F = 16;
  const int Nn = in_sizes[0] / IN_F;          // 100000
  const int E  = in_sizes[1] / 2;             // 1600000

  const float* x        = (const float*)d_in[0];
  const int*   ei       = (const int*)d_in[1];
  const float* W1       = (const float*)d_in[2];
  const float* att_src1 = (const float*)d_in[3];
  const float* att_dst1 = (const float*)d_in[4];
  const float* b1       = (const float*)d_in[5];
  const float* W2       = (const float*)d_in[6];
  const float* att_src2 = (const float*)d_in[7];
  const float* att_dst2 = (const float*)d_in[8];
  const float* b2       = (const float*)d_in[9];
  float* out = (float*)d_out;

  const int* srcv = ei;                       // edge_index[0,:]
  const int* dstv = ei + E;                   // edge_index[1,:]

  // ---- workspace carve-up (256B aligned) ----
  char* p = (char*)d_ws;
  auto alloc = [&](size_t bytes) -> void* {
    void* r = (void*)p; p += (bytes + 255) & ~(size_t)255; return r;
  };
  unsigned short* xb    = (unsigned short*)alloc((size_t)Nn * IN_F * 2);       // 25.6 MB
  unsigned short* w1t   = (unsigned short*)alloc((size_t)IN_F * H * HF * 2);   // [256][128]
  unsigned short* w2t   = (unsigned short*)alloc((size_t)HF * H * OUT_F * 2);  // [128][32]
  float*          h1    = (float*)alloc((size_t)Nn * H * HF * 4);              // 102.4 MB
  float*          h2    = (float*)alloc((size_t)Nn * H * OUT_F * 4);           // 51.2 MB
  float*          a_s   = (float*)alloc((size_t)Nn * H * 4);
  float*          a_d   = (float*)alloc((size_t)Nn * H * 4);
  unsigned*       menc  = (unsigned*)alloc((size_t)Nn * H * 4);
  float*          ssum  = (float*)alloc((size_t)Nn * H * 4);
  float*          anum  = (float*)alloc((size_t)E * H * 4);                    // 51.2 MB
  float*          acc1  = (float*)alloc((size_t)Nn * HF * 4);                  // 12.8 MB
  unsigned short* hmidb = (unsigned short*)alloc((size_t)Nn * HF * 2);

  const int T = 256;
  auto nb = [&](int n) { return (n + T - 1) / T; };
  const int EH = E * H;                       // 12.8M, fits in int

  // ---- cast x to bf16; transpose+cast weights to Bt[N][K] bf16 ----
  k_f32_to_bf16<<<nb(Nn * IN_F), T, 0, stream>>>(x, xb, Nn * IN_F);
  k_transpose_f32_to_bf16<<<nb(IN_F * H * HF), T, 0, stream>>>(W1, w1t, IN_F, H * HF);
  k_transpose_f32_to_bf16<<<nb(HF * H * OUT_F), T, 0, stream>>>(W2, w2t, HF, H * OUT_F);

  // =================== layer 1 ===================
  {
    dim3 blk(32, 8), grd(Nn / 16, (H * HF) / 128);  // N=256 -> gridDim.y=2
    k_wmma_gemm_bf16<128><<<grd, blk, 0, stream>>>(xb, w1t, h1, Nn, H * HF);
  }
  k_att_coef<<<nb(Nn * H), T, 0, stream>>>(h1, att_src1, att_dst1, a_s, a_d, Nn, H, HF);

  k_fill_u32<<<nb(Nn * H), T, 0, stream>>>(menc, ENC_NEG_INF_U32, Nn * H);
  k_fill_u32<<<nb(Nn * H), T, 0, stream>>>((unsigned*)ssum, 0u, Nn * H);
  k_fill_u32<<<nb(Nn * HF), T, 0, stream>>>((unsigned*)acc1, 0u, Nn * HF);

  k_edge_max<<<nb(EH), T, 0, stream>>>(srcv, dstv, a_s, a_d, menc, E, H);
  k_edge_expsum<<<nb(EH), T, 0, stream>>>(srcv, dstv, a_s, a_d, menc, anum, ssum, E, H);
  k_message<<<nb(EH), T, 0, stream>>>(srcv, dstv, h1, anum, ssum, acc1, E, H, HF, 1.0f / H);

  k_finalize1<<<nb(Nn * HF), T, 0, stream>>>(acc1, b1, hmidb, Nn, HF);

  // =================== layer 2 ===================
  {
    dim3 blk(32, 8), grd(Nn / 16, (H * OUT_F) / 128);  // N=128 -> gridDim.y=1
    k_wmma_gemm_bf16<32><<<grd, blk, 0, stream>>>(hmidb, w2t, h2, Nn, H * OUT_F);
  }
  k_att_coef<<<nb(Nn * H), T, 0, stream>>>(h2, att_src2, att_dst2, a_s, a_d, Nn, H, OUT_F);

  k_fill_u32<<<nb(Nn * H), T, 0, stream>>>(menc, ENC_NEG_INF_U32, Nn * H);
  k_fill_u32<<<nb(Nn * H), T, 0, stream>>>((unsigned*)ssum, 0u, Nn * H);
  k_fill_u32<<<nb(Nn * OUT_F), T, 0, stream>>>((unsigned*)out, 0u, Nn * OUT_F);

  k_edge_max<<<nb(EH), T, 0, stream>>>(srcv, dstv, a_s, a_d, menc, E, H);
  k_edge_expsum<<<nb(EH), T, 0, stream>>>(srcv, dstv, a_s, a_d, menc, anum, ssum, E, H);
  k_message<<<nb(EH), T, 0, stream>>>(srcv, dstv, h2, anum, ssum, out, E, H, OUT_F, 1.0f / H);

  k_finalize2<<<nb(Nn * OUT_F), T, 0, stream>>>(out, b2, Nn, OUT_F);
}
